// MaceBlock_79508434583945
// MI455X (gfx1250) — matrix operand order repaired and checked by reading, not compiled, not run on hardware
//
#include <hip/hip_runtime.h>
#include <hip/hip_bf16.h>
#include <math.h>

// ---------------------------------------------------------------------------
// MACE-like block for MI455X (gfx1250, wave32, WMMA).
//   k_init       : zero agg1/agg2/out
//   k_packB/A    : pre-swizzle static weights into WMMA f16 register layout
//   k_edge       : r, u, sh[16], bessel[8] per edge
//   k_lin_gather : h1 = embed[spec] @ l1_lin_up          (WMMA, 16-node tiles)
//   k_msg1       : rw1 = silu(rad@W1)@W2 (WMMA) fused atomic scatter
//   k_feats      : feats = selW[spec] @ agg1 (WMMA) + pointwise -> out0,outv,ro1
//   k_lin_plain  : h2 = out0 @ l2_lin_up                 (WMMA)
//   k_msg2       : rw2 col-0 GEMM (WMMA) fused edge_dot + atomic scatter
//   k_final      : res matvec + poly + readout MLP -> ro2
// ---------------------------------------------------------------------------

typedef __attribute__((ext_vector_type(16))) _Float16 v16h;
typedef __attribute__((ext_vector_type(8)))  float    v8f;

#define CC 64
#define NBQ 8
#define HH 64
#define R_MAX 5.0f
#define AVG_INV 0.1f

__device__ __forceinline__ void atomic_add_f32(float* p, float v) {
    __hip_atomic_fetch_add(p, v, __ATOMIC_RELAXED, __HIP_MEMORY_SCOPE_AGENT);
}

__device__ __forceinline__ float silu(float x) {
    return x / (1.0f + __expf(-x));
}

// A-matrix 16x32 f16 layout (ISA 7.12.2): lane m=L%16, half=L/16,
// VGPR j<4 -> K=2j,2j+1 (+8*half), j>=4 -> K=16+2(j-4) (+8*half)
__device__ __forceinline__ int a_kbase(int j, int half) {
    return ((j < 4) ? (2 * j) : (16 + 2 * (j - 4))) + half * 8;
}

__device__ __forceinline__ v16h load_a_f32(const float* src, int stride, int kOff) {
    int lane = threadIdx.x & 31;
    int m = lane & 15, half = lane >> 4;
    v16h a;
#pragma unroll
    for (int j = 0; j < 8; ++j) {
        int kb = a_kbase(j, half) + kOff;
        a[2 * j]     = (_Float16)src[m * stride + kb];
        a[2 * j + 1] = (_Float16)src[m * stride + kb + 1];
    }
    return a;
}

__device__ __forceinline__ v16h load_a_lds(const _Float16* src, int stride, int kOff) {
    int lane = threadIdx.x & 31;
    int m = lane & 15, half = lane >> 4;
    v16h a;
#pragma unroll
    for (int j = 0; j < 8; ++j) {
        int kb = a_kbase(j, half) + kOff;
        a[2 * j]     = src[m * stride + kb];
        a[2 * j + 1] = src[m * stride + kb + 1];
    }
    return a;
}

__device__ __forceinline__ v16h load_a_gather(const float* W, const int* spec,
                                              int n0, int N, int stride, int kOff) {
    int lane = threadIdx.x & 31;
    int m = lane & 15, half = lane >> 4;
    int node = n0 + m; if (node > N - 1) node = N - 1;
    const float* row = W + (size_t)spec[node] * stride;
    v16h a;
#pragma unroll
    for (int j = 0; j < 8; ++j) {
        int kb = a_kbase(j, half) + kOff;
        a[2 * j]     = (_Float16)row[kb];
        a[2 * j + 1] = (_Float16)row[kb + 1];
    }
    return a;
}

// B-matrix 32x16 f16 layout: same K order as A (VGPR j -> K=a_kbase(j,half),+1).
// Element B[k][(col0+n)*colstep].
__device__ __forceinline__ v16h load_b_f32(const float* src, int stride, int kOff,
                                           int col0, int colstep) {
    int lane = threadIdx.x & 31;
    int n = lane & 15, half = lane >> 4;
    int col = (col0 + n) * colstep;
    v16h b;
#pragma unroll
    for (int j = 0; j < 8; ++j) {
        int k = kOff + a_kbase(j, half);
        b[2 * j]     = (_Float16)src[(size_t)k * stride + col];
        b[2 * j + 1] = (_Float16)src[(size_t)(k + 1) * stride + col];
    }
    return b;
}

// Pre-packed fragment fetch: one 32-byte load per lane per WMMA operand.
__device__ __forceinline__ v16h load_pk(const _Float16* pk, int frag) {
    int lane = threadIdx.x & 31;
    return *(const v16h*)(pk + ((size_t)frag * 32 + lane) * 16);
}

__device__ __forceinline__ v8f wmma16(v16h a, v16h b, v8f c) {
    return __builtin_amdgcn_wmma_f32_16x16x32_f16(false, a, false, b, (short)0, c,
                                                  false, false);
}

// ---------------------------------------------------------------------------
__global__ void k_init(float* agg1, float* agg2, float* out, int n1, int n2, int n3) {
    int total = n1 + n2 + n3;
    for (int i = blockIdx.x * blockDim.x + threadIdx.x; i < total;
         i += gridDim.x * blockDim.x) {
        if (i < n1) agg1[i] = 0.0f;
        else if (i < n1 + n2) agg2[i - n1] = 0.0f;
        else out[i - n1 - n2] = 0.0f;
    }
}

// Pack B-matrix tiles into WMMA register layout: frag index = tile*2 + kchunk,
// 32 lanes x 16 halfs per frag.
__global__ void k_packB(const float* __restrict__ src, int stride, int ntiles,
                        int colstep, _Float16* __restrict__ dst) {
    int gid = blockIdx.x * blockDim.x + threadIdx.x;   // ntiles*2*32 threads
    if (gid >= ntiles * 64) return;
    int lane = gid & 31, kc = (gid >> 5) & 1, t = gid >> 6;
    int n = lane & 15, half = lane >> 4;
    int col = (t * 16 + n) * colstep;
    _Float16* d = dst + (size_t)gid * 16;
#pragma unroll
    for (int j = 0; j < 8; ++j) {
        int k = kc * 32 + a_kbase(j, half);
        d[2 * j]     = (_Float16)src[(size_t)k * stride + col];
        d[2 * j + 1] = (_Float16)src[(size_t)(k + 1) * stride + col];
    }
}

// Pack l1_sel_W (S x 64 x 64) A-matrix tiles: frag = ((sp*4 + ct)*2 + kc).
__global__ void k_packA_sel(const float* __restrict__ selW, int S,
                            _Float16* __restrict__ dst) {
    int gid = blockIdx.x * blockDim.x + threadIdx.x;   // S*4*2*32 threads
    if (gid >= S * 256) return;
    int lane = gid & 31, kc = (gid >> 5) & 1, ct = (gid >> 6) & 3, sp = gid >> 8;
    int m = lane & 15, half = lane >> 4;
    const float* row = selW + (size_t)sp * CC * CC + (size_t)(ct * 16 + m) * CC;
    _Float16* d = dst + (size_t)gid * 16;
#pragma unroll
    for (int j = 0; j < 8; ++j) {
        int kb = a_kbase(j, half) + kc * 32;
        d[2 * j]     = (_Float16)row[kb];
        d[2 * j + 1] = (_Float16)row[kb + 1];
    }
}

// ---------------------------------------------------------------------------
__global__ void k_edge(const float* __restrict__ ev, float* __restrict__ uo,
                       float* __restrict__ sho, float* __restrict__ rado, int E) {
    int e = blockIdx.x * blockDim.x + threadIdx.x;
    if (e >= E) return;
    float x = ev[e * 3 + 0], y = ev[e * 3 + 1], z = ev[e * 3 + 2];
    float r = sqrtf(x * x + y * y + z * z + 1e-12f);
    float inv = 1.0f / r;
    x *= inv; y *= inv; z *= inv;
    uo[e * 3 + 0] = x; uo[e * 3 + 1] = y; uo[e * 3 + 2] = z;

    const float s3 = 1.7320508075688772f, s5 = 2.23606797749979f;
    const float s15 = 3.872983346207417f, s7 = 2.6457513110645907f;
    const float s105 = 10.246950765959598f;
    const float ca = 2.0916500663351889f;   // sqrt(35/8)
    const float cb = 1.6201851746019651f;   // sqrt(21/8)
    float* sh = sho + (size_t)e * 16;
    sh[0]  = 1.0f;
    sh[1]  = s3 * x;
    sh[2]  = s3 * y;
    sh[3]  = s3 * z;
    sh[4]  = s15 * x * y;
    sh[5]  = s15 * y * z;
    sh[6]  = 0.5f * s5 * (3.0f * z * z - 1.0f);
    sh[7]  = s15 * x * z;
    sh[8]  = 0.5f * s15 * (x * x - y * y);
    sh[9]  = ca * y * (3.0f * x * x - y * y);
    sh[10] = s105 * x * y * z;
    sh[11] = cb * y * (5.0f * z * z - 1.0f);
    sh[12] = 0.5f * s7 * z * (5.0f * z * z - 3.0f);
    sh[13] = cb * x * (5.0f * z * z - 1.0f);
    sh[14] = 0.5f * s105 * z * (x * x - y * y);
    sh[15] = ca * x * (x * x - 3.0f * y * y);

    float ur = r * (1.0f / R_MAX);
    float env = (ur < 1.0f) ? (1.0f - ur) * (1.0f - ur) * (1.0f + 2.0f * ur) : 0.0f;
    float cc = 0.6324555320336759f * env * inv;  // sqrt(2/R_MAX)*env/r
    const float PI = 3.14159265358979323846f;
#pragma unroll
    for (int k = 1; k <= NBQ; ++k)
        rado[(size_t)e * 8 + (k - 1)] = cc * __sinf((float)k * PI * ur);
}

// ---------------------------------------------------------------------------
// h = A_rows @ lin. One wave / 16 rows. lin pre-packed (4 tiles x 2 kchunks).
__global__ void k_lin_gather(const float* __restrict__ W, const int* __restrict__ spec,
                             const _Float16* __restrict__ pkLin,
                             float* __restrict__ out, int N) {
    int n0 = blockIdx.x * 16;
    int lane = threadIdx.x & 31;
    bool full = (n0 + 16 <= N);
    v16h a0 = load_a_gather(W, spec, n0, N, CC, 0);
    v16h a1 = load_a_gather(W, spec, n0, N, CC, 32);
    int n = lane & 15, half = lane >> 4;
#pragma unroll
    for (int t = 0; t < 4; ++t) {
        v16h b0 = load_pk(pkLin, t * 2 + 0);
        v16h b1 = load_pk(pkLin, t * 2 + 1);
        v8f c = {};
        c = wmma16(a0, b0, c);
        c = wmma16(a1, b1, c);
        if (full) {
#pragma unroll
            for (int i = 0; i < 8; ++i)
                out[(size_t)(n0 + i + 8 * half) * CC + t * 16 + n] = c[i];
        } else {
#pragma unroll
            for (int i = 0; i < 8; ++i) {
                int row = n0 + i + 8 * half;
                if (row < N) out[(size_t)row * CC + t * 16 + n] = c[i];
            }
        }
    }
}

__global__ void k_lin_plain(const float* __restrict__ X,
                            const _Float16* __restrict__ pkLin,
                            float* __restrict__ out, int N) {
    int n0 = blockIdx.x * 16;
    int lane = threadIdx.x & 31;
    bool full = (n0 + 16 <= N);
    v16h a0 = load_a_f32(X + (size_t)n0 * CC, CC, 0);
    v16h a1 = load_a_f32(X + (size_t)n0 * CC, CC, 32);
    int n = lane & 15, half = lane >> 4;
#pragma unroll
    for (int t = 0; t < 4; ++t) {
        v16h b0 = load_pk(pkLin, t * 2 + 0);
        v16h b1 = load_pk(pkLin, t * 2 + 1);
        v8f c = {};
        c = wmma16(a0, b0, c);
        c = wmma16(a1, b1, c);
        if (full) {
#pragma unroll
            for (int i = 0; i < 8; ++i)
                out[(size_t)(n0 + i + 8 * half) * CC + t * 16 + n] = c[i];
        } else {
#pragma unroll
            for (int i = 0; i < 8; ++i) {
                int row = n0 + i + 8 * half;
                if (row < N) out[(size_t)row * CC + t * 16 + n] = c[i];
            }
        }
    }
}

// ---------------------------------------------------------------------------
// Layer-1 message: rw1 = silu(rad@rW1)@rW2 via WMMA, fused with
//   agg1[r, c, k] += rw1[e, c*4+l] * sh[e,k] * h1[snd,c]   (L_IDX[k]==l)
// One wave per 16-edge tile; LDS stages silu activations; rW2 pre-packed.
__global__ void k_msg1(const float* __restrict__ rad, const float* __restrict__ sh,
                       const float* __restrict__ rW1, const _Float16* __restrict__ pkW2,
                       const float* __restrict__ h1, const int* __restrict__ snd,
                       const int* __restrict__ rcv, float* __restrict__ agg1, int E) {
    __shared__ _Float16 tb[16 * HH];
    int e0 = blockIdx.x * 16;
    int lane = threadIdx.x & 31;
    bool full = (e0 + 16 <= E);
    // stage 1: t = silu(rad @ rW1), 16 edges x 64
    for (int idx = lane; idx < 16 * HH; idx += 32) {
        int le = idx >> 6, hc = idx & 63;
        int e = e0 + le; if (e > E - 1) e = E - 1;
        const float* rp = rad + (size_t)e * 8;
        float acc = 0.0f;
#pragma unroll
        for (int kk = 0; kk < NBQ; ++kk) acc += rp[kk] * rW1[kk * HH + hc];
        tb[idx] = (_Float16)silu(acc);
    }
    asm volatile("s_wait_dscnt 0" ::: "memory");

    v16h a0 = load_a_lds(tb, HH, 0);
    v16h a1 = load_a_lds(tb, HH, 32);
    int nlo = lane & 15, half = lane >> 4;
    // preload per-lane edge indices (8 edges per lane half)
    int se[8], re[8];
#pragma unroll
    for (int i = 0; i < 8; ++i) {
        int e = e0 + i + 8 * half; if (e > E - 1) e = E - 1;
        se[i] = snd[e]; re[i] = rcv[e];
    }
    const int KS[4] = {0, 1, 4, 9};
    const int KE[4] = {1, 4, 9, 16};
#pragma unroll 1
    for (int t = 0; t < 16; ++t) {
        v16h b0 = load_pk(pkW2, t * 2 + 0);
        v16h b1 = load_pk(pkW2, t * 2 + 1);
        v8f c = {};
        c = wmma16(a0, b0, c);
        c = wmma16(a1, b1, c);
        int col = t * 16 + nlo;        // flattened (c,4) index
        int cch = col >> 2;            // channel
        int l = col & 3;               // l component
        int ks = KS[l], ke = KE[l];
#pragma unroll
        for (int i = 0; i < 8; ++i) {
            int e = e0 + i + 8 * half;
            if (full || e < E) {
                float base = c[i] * h1[(size_t)se[i] * CC + cch];
                float* ap = agg1 + (size_t)re[i] * (CC * 16) + cch * 16;
                const float* sp = sh + (size_t)e * 16;
                for (int k = ks; k < ke; ++k) atomic_add_f32(ap + k, base * sp[k]);
            }
        }
    }
}

// ---------------------------------------------------------------------------
// feats = selW[spec[n]] @ agg1[n]  (64x64 @ 64x16) + pointwise nonlinearity.
// One wave per node; selW pre-packed into A fragments.
__global__ void k_feats(const float* __restrict__ agg1, const int* __restrict__ spec,
                        const _Float16* __restrict__ pkSel, const float* __restrict__ pwT,
                        const float* __restrict__ uwT, const float* __restrict__ row,
                        float* __restrict__ out0, float* __restrict__ outv,
                        float* __restrict__ out, int N) {
    __shared__ float fe[CC * 16];
    int nn = blockIdx.x;
    int lane = threadIdx.x & 31;
    int sp = spec[nn];
    const float* B = agg1 + (size_t)nn * (CC * 16);
    v16h b0 = load_b_f32(B, 16, 0, 0, 1);
    v16h b1 = load_b_f32(B, 16, 32, 0, 1);
    int nlo = lane & 15, half = lane >> 4;
#pragma unroll
    for (int t = 0; t < 4; ++t) {
        v16h a0 = load_pk(pkSel, (sp * 4 + t) * 2 + 0);
        v16h a1 = load_pk(pkSel, (sp * 4 + t) * 2 + 1);
        v8f c = {};
        c = wmma16(a0, b0, c);
        c = wmma16(a1, b1, c);
#pragma unroll
        for (int i = 0; i < 8; ++i)
            fe[(t * 16 + i + 8 * half) * 16 + nlo] = c[i] * AVG_INV;
    }
    asm volatile("s_wait_dscnt 0" ::: "memory");

    const float* pw = pwT + (size_t)sp * 3 * CC;
    const float* uw = uwT + (size_t)sp * 3 * CC;
    float acc = 0.0f;
    for (int c0 = lane; c0 < CC; c0 += 32) {
        float s  = fe[c0 * 16 + 0];
        float v0 = fe[c0 * 16 + 1], v1 = fe[c0 * 16 + 2], v2 = fe[c0 * 16 + 3];
        float o = pw[c0] * s + pw[CC + c0] * s * s + pw[2 * CC + c0] * s * s * s;
        float f = uw[c0] + uw[CC + c0] * s + uw[2 * CC + c0] * s * s;
        out0[(size_t)nn * CC + c0] = o;
        outv[(size_t)nn * CC * 3 + c0 * 3 + 0] = f * v0;
        outv[(size_t)nn * CC * 3 + c0 * 3 + 1] = f * v1;
        outv[(size_t)nn * CC * 3 + c0 * 3 + 2] = f * v2;
        acc += o * row[c0];
    }
#pragma unroll
    for (int off = 16; off > 0; off >>= 1) acc += __shfl_xor(acc, off, 32);
    if (lane == 0) out[nn * 2 + 0] = acc;
}

// ---------------------------------------------------------------------------
// Layer-2 message: only k=0 survives (sh[:,0]==1, L_IDX[0]==0), so the radial
// GEMM shrinks to columns 4c of l2_rW2 (pre-packed). WMMA + fused edge-dot +
// atomic scatter.
__global__ void k_msg2(const float* __restrict__ rad, const float* __restrict__ u,
                       const float* __restrict__ rW1, const _Float16* __restrict__ pkW2,
                       const float* __restrict__ h2, const float* __restrict__ outv,
                       const int* __restrict__ snd, const int* __restrict__ rcv,
                       float* __restrict__ agg2, int E) {
    __shared__ _Float16 tb[16 * HH];
    int e0 = blockIdx.x * 16;
    int lane = threadIdx.x & 31;
    bool full = (e0 + 16 <= E);
    for (int idx = lane; idx < 16 * HH; idx += 32) {
        int le = idx >> 6, hc = idx & 63;
        int e = e0 + le; if (e > E - 1) e = E - 1;
        const float* rp = rad + (size_t)e * 8;
        float acc = 0.0f;
#pragma unroll
        for (int kk = 0; kk < NBQ; ++kk) acc += rp[kk] * rW1[kk * HH + hc];
        tb[idx] = (_Float16)silu(acc);
    }
    asm volatile("s_wait_dscnt 0" ::: "memory");

    v16h a0 = load_a_lds(tb, HH, 0);
    v16h a1 = load_a_lds(tb, HH, 32);
    int nlo = lane & 15, half = lane >> 4;
    int se[8], re[8];
#pragma unroll
    for (int i = 0; i < 8; ++i) {
        int e = e0 + i + 8 * half; if (e > E - 1) e = E - 1;
        se[i] = snd[e]; re[i] = rcv[e];
    }
#pragma unroll 1
    for (int t = 0; t < 4; ++t) {
        v16h b0 = load_pk(pkW2, t * 2 + 0);
        v16h b1 = load_pk(pkW2, t * 2 + 1);
        v8f c = {};
        c = wmma16(a0, b0, c);
        c = wmma16(a1, b1, c);
        int cch = t * 16 + nlo;
#pragma unroll
        for (int i = 0; i < 8; ++i) {
            int e = e0 + i + 8 * half;
            if (full || e < E) {
                const float* ov = outv + (size_t)se[i] * CC * 3 + cch * 3;
                const float* uu = u + (size_t)e * 3;
                float dot = ov[0] * uu[0] + ov[1] * uu[1] + ov[2] * uu[2];
                float val = c[i] * (h2[(size_t)se[i] * CC + cch] + dot);
                atomic_add_f32(agg2 + (size_t)re[i] * CC + cch, val);
            }
        }
    }
}

// ---------------------------------------------------------------------------
// res matvec + cubic poly + 64->16 silu readout. One wave per node.
__global__ void k_final(const float* __restrict__ agg2, const int* __restrict__ spec,
                        const float* __restrict__ resW, const float* __restrict__ out0,
                        const float* __restrict__ p2T, const float* __restrict__ roW1,
                        const float* __restrict__ roW2, float* __restrict__ out, int N) {
    __shared__ float so0[CC];
    __shared__ float so2[CC];
    int nn = blockIdx.x;
    int lane = threadIdx.x & 31;
    int sp = spec[nn];
    for (int c0 = lane; c0 < CC; c0 += 32) so0[c0] = out0[(size_t)nn * CC + c0];
    asm volatile("s_wait_dscnt 0" ::: "memory");
    const float* p2 = p2T + (size_t)sp * 3 * CC;
    for (int c0 = lane; c0 < CC; c0 += 32) {
        const float* Wr = resW + (size_t)sp * CC * CC + (size_t)c0 * CC;
        float res = 0.0f;
#pragma unroll
        for (int d = 0; d < CC; ++d) res += Wr[d] * so0[d];
        float s2 = agg2[(size_t)nn * CC + c0] * AVG_INV;
        float o2 = p2[c0] * s2 + p2[CC + c0] * s2 * s2 + p2[2 * CC + c0] * s2 * s2 * s2
                 + res;
        so2[c0] = o2;
    }
    asm volatile("s_wait_dscnt 0" ::: "memory");
    float contrib = 0.0f;
    if (lane < 16) {
        float tj = 0.0f;
#pragma unroll
        for (int c0 = 0; c0 < CC; ++c0) tj += so2[c0] * roW1[c0 * 16 + lane];
        contrib = silu(tj) * roW2[lane];
    }
#pragma unroll
    for (int off = 16; off > 0; off >>= 1) contrib += __shfl_xor(contrib, off, 32);
    if (lane == 0) out[nn * 2 + 1] = contrib;
}

// ---------------------------------------------------------------------------
extern "C" void kernel_launch(void* const* d_in, const int* in_sizes, int n_in,
                              void* d_out, int out_size, void* d_ws, size_t ws_size,
                              hipStream_t stream) {
    const float* ev      = (const float*)d_in[0];
    const int*   spec    = (const int*)d_in[1];
    const int*   snd     = (const int*)d_in[2];
    const int*   rcv     = (const int*)d_in[3];
    const float* embedW  = (const float*)d_in[4];
    const float* l1_lin  = (const float*)d_in[5];
    const float* l1_rW1  = (const float*)d_in[6];
    const float* l1_rW2  = (const float*)d_in[7];
    const float* l1_selW = (const float*)d_in[8];
    const float* l1_pw   = (const float*)d_in[9];
    const float* l1_uw   = (const float*)d_in[10];
    const float* l1_row  = (const float*)d_in[11];
    const float* l2_lin  = (const float*)d_in[12];
    const float* l2_rW1  = (const float*)d_in[13];
    const float* l2_rW2  = (const float*)d_in[14];
    const float* l2_resW = (const float*)d_in[15];
    const float* l2_pw   = (const float*)d_in[16];
    const float* l2_roW1 = (const float*)d_in[17];
    const float* l2_roW2 = (const float*)d_in[18];
    float* out = (float*)d_out;

    const int E = in_sizes[0] / 3;
    const int N = in_sizes[1];
    const int S = in_sizes[4] / CC;   // species count

    float* ws     = (float*)d_ws;
    float* w_u    = ws;                        // 3E
    float* w_sh   = w_u + (size_t)3 * E;       // 16E
    float* w_rad  = w_sh + (size_t)16 * E;     // 8E
    float* w_h1   = w_rad + (size_t)8 * E;     // 64N
    float* w_agg1 = w_h1 + (size_t)64 * N;     // 1024N
    float* w_out0 = w_agg1 + (size_t)1024 * N; // 64N
    float* w_outv = w_out0 + (size_t)64 * N;   // 192N
    float* w_h2   = w_outv + (size_t)192 * N;  // 64N
    float* w_agg2 = w_h2 + (size_t)64 * N;     // 64N

    // f16 packed-weight region (after float region)
    _Float16* hb       = (_Float16*)(w_agg2 + (size_t)64 * N);
    _Float16* p_l1rw2  = hb;                   // 16 tiles * 2 * 32 * 16 = 16384
    _Float16* p_l2rw2  = p_l1rw2 + 16384;      // 4 tiles -> 4096
    _Float16* p_l1lin  = p_l2rw2 + 4096;       // 4096
    _Float16* p_l2lin  = p_l1lin + 4096;       // 4096
    _Float16* p_selA   = p_l2lin + 4096;       // S*4*2*32*16

    const int ntile = (N + 15) / 16;
    const int etile = (E + 15) / 16;

    k_init<<<2048, 256, 0, stream>>>(w_agg1, w_agg2, out, N * 1024, N * 64, N * 2);
    k_packB<<<(16 * 64 + 255) / 256, 256, 0, stream>>>(l1_rW2, 256, 16, 1, p_l1rw2);
    k_packB<<<1, 256, 0, stream>>>(l2_rW2, 256, 4, 4, p_l2rw2);
    k_packB<<<1, 256, 0, stream>>>(l1_lin, CC, 4, 1, p_l1lin);
    k_packB<<<1, 256, 0, stream>>>(l2_lin, CC, 4, 1, p_l2lin);
    k_packA_sel<<<(S * 256 + 255) / 256, 256, 0, stream>>>(l1_selW, S, p_selA);

    k_edge<<<(E + 255) / 256, 256, 0, stream>>>(ev, w_u, w_sh, w_rad, E);
    k_lin_gather<<<ntile, 32, 0, stream>>>(embedW, spec, p_l1lin, w_h1, N);
    k_msg1<<<etile, 32, 0, stream>>>(w_rad, w_sh, l1_rW1, p_l1rw2, w_h1, snd, rcv,
                                     w_agg1, E);
    k_feats<<<N, 32, 0, stream>>>(w_agg1, spec, p_selA, l1_pw, l1_uw, l1_row,
                                  w_out0, w_outv, out, N);
    k_lin_plain<<<ntile, 32, 0, stream>>>(w_out0, p_l2lin, w_h2, N);
    k_msg2<<<etile, 32, 0, stream>>>(w_rad, w_u, l2_rW1, p_l2rw2, w_h2, w_outv, snd,
                                     rcv, w_agg2, E);
    k_final<<<N, 32, 0, stream>>>(w_agg2, spec, l2_resW, w_out0, l2_pw, l2_roW1,
                                  l2_roW2, out, N);
}
